// FusedMoE_42262478192977
// MI455X (gfx1250) — compile-verified, matrix operand order checked by reading, not run on hardware
//
#include <hip/hip_runtime.h>
#include <math.h>

// Problem constants (from reference setup_inputs)
#define T_TOK 2048
#define HDIM  1024
#define IDIM  1024
#define NEXP  8
#define N2    (2 * IDIM)   // 2048 columns of w13
#define KT13  (HDIM / 32)  // 32 K-tiles for GEMM1
#define KT2   (IDIM / 32)  // 32 K-tiles for GEMM2

typedef __bf16 bf16;
typedef bf16  v16bf __attribute__((ext_vector_type(16)));
typedef float v8f   __attribute__((ext_vector_type(8)));

#if __has_builtin(__builtin_amdgcn_global_load_async_to_lds_b128) && \
    __has_builtin(__builtin_amdgcn_s_wait_asynccnt)
#define MOE_HAVE_ASYNC 1
// exact parameter type per hipcc diagnostic: int vector_size(16), AS1 src / AS3 dst
typedef int vi4 __attribute__((vector_size(16)));
typedef __attribute__((address_space(1))) vi4 gvi4;
typedef __attribute__((address_space(3))) vi4 lvi4;
#endif

// fp32 -> bf16 round-to-nearest-even (scalar fallback)
__device__ __forceinline__ unsigned short f2bf(float f) {
    unsigned u = __builtin_bit_cast(unsigned, f);
    u += 0x7FFFu + ((u >> 16) & 1u);
    return (unsigned short)(u >> 16);
}

// two fp32 -> packed bf16 dword; prefer HW v_cvt_pk_bf16_f32 on gfx1250
__device__ __forceinline__ unsigned pk2bf(float a, float b) {
#if __has_builtin(__builtin_amdgcn_cvt_pk_bf16_f32)
    auto r = __builtin_amdgcn_cvt_pk_bf16_f32(a, b);
    return __builtin_bit_cast(unsigned, r);
#else
    return (unsigned)f2bf(a) | ((unsigned)f2bf(b) << 16);
#endif
}

// 16 bf16 lanes-worth of a WMMA operand fragment (32 bytes)
union BFrag {
    unsigned short s[16];
    uint4 q[2];
    v16bf v;
};

#define WMMA_BF16(A, B, C) __builtin_amdgcn_wmma_f32_16x16x32_bf16( \
    false, (A), false, (B), (short)0, (C), false, false)

// ---------------------------------------------------------------------------
// Kernel 0: zero output (atomic accumulation target) and expert counters
// ---------------------------------------------------------------------------
__global__ void moe_zero(float* __restrict__ out, int* __restrict__ counts) {
    int i = blockIdx.x * blockDim.x + threadIdx.x;
    if (i < T_TOK * HDIM) out[i] = 0.0f;
    if (i < NEXP) counts[i] = 0;
}

// ---------------------------------------------------------------------------
// Kernel 1a: x fp32 -> bf16 (flat layout, row-major [T][H])
// ---------------------------------------------------------------------------
__global__ void moe_conv_x(const float* __restrict__ src, unsigned* __restrict__ dst,
                           int ndw) {
    int i = blockIdx.x * blockDim.x + threadIdx.x;
    if (i >= ndw) return;
    float2 v = ((const float2*)src)[i];
    dst[i] = pk2bf(v.x, v.y);
}

// ---------------------------------------------------------------------------
// Kernel 1b: weight fp32 [1024][N] -> bf16 in WMMA B-fragment lane layout.
// Fragment (kt,ntile): lane l holds col n=ntile*16+(l&15),
//   k = kt*32 + (l>>4)*8 + {0..7} (s[0..7]) and +16+{0..7} (s[8..15]).
// Packed address (bf16 elems): ((ntile*32 + kt)*32 + l)*16 + j  -> 32B/lane,
// so the consumer reads each fragment as two coalesced b128 loads.
// One thread converts one dword (j-pair); pairs never straddle the j=8 split.
// ---------------------------------------------------------------------------
__global__ void moe_conv_w(const float* __restrict__ srcAll, unsigned* __restrict__ dstAll,
                           int N) {
    const int K = 1024;                 // both weight matrices have K = 1024
    const int e = blockIdx.y;
    const float* src = srcAll + (size_t)e * K * N;
    unsigned* dst = dstAll + (size_t)e * ((size_t)K * N >> 1);

    int idx = blockIdx.x * blockDim.x + threadIdx.x;   // dword index
    if (idx >= (K * N) >> 1) return;

    int j2   = idx & 7;            // dword within lane's 16-elem fragment
    int l    = (idx >> 3) & 31;    // lane
    int rest = idx >> 8;           // ntile*32 + kt
    int kt   = rest & 31;
    int nt   = rest >> 5;
    int j    = j2 << 1;
    int kof  = (j < 8) ? j : (j + 8);
    int k    = (kt << 5) + ((l >> 4) << 3) + kof;
    int n    = (nt << 4) + (l & 15);
    float a = src[(size_t)k * N + n];
    float b = src[(size_t)(k + 1) * N + n];
    dst[idx] = pk2bf(a, b);
}

// ---------------------------------------------------------------------------
// Kernel 2: softmax + top-2 + renormalize; scatter tokens into expert buckets
// ---------------------------------------------------------------------------
__global__ void moe_route(const float* __restrict__ logits,
                          int* __restrict__ counts,
                          int* __restrict__ btok,
                          float* __restrict__ bw) {
    int t = blockIdx.x * blockDim.x + threadIdx.x;
    if (t >= T_TOK) return;

    float l[NEXP];
    float m = -3.4e38f;
    #pragma unroll
    for (int e = 0; e < NEXP; ++e) { l[e] = logits[t * NEXP + e]; m = fmaxf(m, l[e]); }
    float p[NEXP]; float s = 0.0f;
    #pragma unroll
    for (int e = 0; e < NEXP; ++e) { p[e] = __expf(l[e] - m); s += p[e]; }

    // top-2 (first occurrence wins ties, matching jax.lax.top_k)
    int i0 = 0;
    #pragma unroll
    for (int e = 1; e < NEXP; ++e) if (p[e] > p[i0]) i0 = e;
    int i1 = (i0 == 0) ? 1 : 0;
    #pragma unroll
    for (int e = 0; e < NEXP; ++e) if (e != i0 && p[e] > p[i1]) i1 = e;

    // renormalized top-2 weights (softmax denominator cancels)
    float inv = __builtin_amdgcn_rcpf(p[i0] + p[i1]);
    float w0 = p[i0] * inv, w1 = p[i1] * inv;

    int s0 = atomicAdd(&counts[i0], 1);
    btok[i0 * T_TOK + s0] = t;  bw[i0 * T_TOK + s0] = w0;
    int s1 = atomicAdd(&counts[i1], 1);
    btok[i1 * T_TOK + s1] = t;  bw[i1 * T_TOK + s1] = w1;
}

// ---------------------------------------------------------------------------
// Kernel 3: grouped expert GEMM, fused SwiGLU, weighted scatter into out.
// Block = 256 threads = 8 waves; each block handles 16 tokens of one expert.
// LDS rows are XOR-swizzled in 16B chunks (chunk ^= row&15) so the 16 lanes
// of each ds_load_b128 spread across all 64 LDS banks. K-loops are manually
// ping-pong double-buffered (two named fragment sets, no register rotation).
// ---------------------------------------------------------------------------
__global__ __launch_bounds__(256) void moe_expert(
    const bf16* __restrict__ xbf, const bf16* __restrict__ w13f,
    const bf16* __restrict__ w2f, const int* __restrict__ counts,
    const int* __restrict__ btok, const float* __restrict__ bw,
    float* __restrict__ out) {
    __shared__ alignas(16) bf16 xs[16 * HDIM];   // gathered x rows (bf16)   32KB
    __shared__ alignas(16) bf16 is[16 * IDIM];   // silu(g)*u intermediate   32KB

    const int e    = blockIdx.y;
    const int tile = blockIdx.x;
    const int cnt  = counts[e];
    const int row0 = tile * 16;
    if (row0 >= cnt) return;                     // tile beyond routed tokens

    const bf16* w13e = w13f + (size_t)e * HDIM * N2;
    const bf16* w2e  = w2f  + (size_t)e * IDIM * HDIM;
    const int*   etok = btok + e * T_TOK;
    const float* ewgt = bw   + e * T_TOK;

    // ---- Phase A: gather 16 pre-converted token rows into LDS (swizzled) --
    {
        const int tid = threadIdx.x;
        const int r   = tid >> 4;               // row 0..15
        const int cb  = (tid & 15) * 64;        // 64 contiguous bf16/thread
        const int gs  = row0 + r;
        const int tok = etok[(gs < cnt) ? gs : row0];
        const uint4* srcq = (const uint4*)(xbf + (size_t)tok * HDIM + cb);
        #pragma unroll
        for (int j = 0; j < 8; ++j) {
            const int chunk = ((cb >> 3) + j) ^ r;          // XOR bank swizzle
            bf16* dst = xs + r * HDIM + (chunk << 3);
#ifdef MOE_HAVE_ASYNC
            __builtin_amdgcn_global_load_async_to_lds_b128(
                (gvi4*)(srcq + j), (lvi4*)dst, 0, 0);
#else
            *(uint4*)dst = srcq[j];
#endif
        }
#ifdef MOE_HAVE_ASYNC
        __builtin_amdgcn_s_wait_asynccnt(0);
#endif
    }
    __syncthreads();

    const int lane = threadIdx.x & 31;
    const int wid  = threadIdx.x >> 5;          // 8 waves
    const int hsel = lane >> 4;                 // half-wave select
    const int ln   = lane & 15;                 // A row / B column sublane

    // ---- Phase B: gu = x @ w13e ; inter = silu(g)*u -> LDS ---------------
    {
        const int tb = wid * 8;                 // first gate column-tile
        #pragma unroll 1
        for (int nt = 0; nt < 8; ++nt) {
            const int ntg = tb + nt;            // gate tile  [0,64)
            const int ntu = ntg + 64;           // up tile    [64,128)
            const bf16* pg = w13e + ((size_t)(ntg * KT13) * 32 + lane) * 16;
            const bf16* pu = w13e + ((size_t)(ntu * KT13) * 32 + lane) * 16;
            const bf16* arow = xs + ln * HDIM;
            v8f accg = {}; v8f accu = {};

            auto loadA = [&](BFrag& A, int kt) {
                const int c0 = kt * 4 + hsel;
                A.q[0] = *(const uint4*)(arow + (((c0)     ^ ln) << 3));
                A.q[1] = *(const uint4*)(arow + (((c0 + 2) ^ ln) << 3));
            };
            auto loadB = [&](BFrag& B, const bf16*& p) {
                B.q[0] = *(const uint4*)(p);
                B.q[1] = *(const uint4*)(p + 8);
                p += 512;
            };

            BFrag A0, Bg0, Bu0, A1, Bg1, Bu1;
            loadA(A0, 0); loadB(Bg0, pg); loadB(Bu0, pu);
            #pragma unroll 1
            for (int kt = 0; kt < KT13 - 2; kt += 2) {
                loadA(A1, kt + 1); loadB(Bg1, pg); loadB(Bu1, pu);
                accg = WMMA_BF16(A0.v, Bg0.v, accg);
                accu = WMMA_BF16(A0.v, Bu0.v, accu);
                loadA(A0, kt + 2); loadB(Bg0, pg); loadB(Bu0, pu);
                accg = WMMA_BF16(A1.v, Bg1.v, accg);
                accu = WMMA_BF16(A1.v, Bu1.v, accu);
            }
            loadA(A1, KT13 - 1); loadB(Bg1, pg); loadB(Bu1, pu);
            accg = WMMA_BF16(A0.v, Bg0.v, accg);
            accu = WMMA_BF16(A0.v, Bu0.v, accu);
            accg = WMMA_BF16(A1.v, Bg1.v, accg);
            accu = WMMA_BF16(A1.v, Bu1.v, accu);

            // SwiGLU, write bf16 intermediate (C/D layout: col=ln, row=r+8*hsel)
            const int cg = ntg * 16 + ln;
            const int cchunk = cg >> 3, csub = cg & 7;
            #pragma unroll
            for (int r = 0; r < 8; ++r) {
                const int m = r + hsel * 8;
                float g = accg[r], u = accu[r];
                float sig = __builtin_amdgcn_rcpf(1.0f + __expf(-g));
                ((unsigned short*)is)[m * IDIM + (((cchunk ^ m) << 3) + csub)] =
                    f2bf(g * sig * u);
            }
        }
    }
    __syncthreads();

    // ---- Phase C: out_rows = inter @ w2e, scale, atomic scatter ----------
    {
        float sc[8]; int tk[8];
        #pragma unroll
        for (int r = 0; r < 8; ++r) {
            const int gs = row0 + r + hsel * 8;
            if (gs < cnt) { sc[r] = ewgt[gs]; tk[r] = etok[gs]; }
            else          { sc[r] = 0.0f;     tk[r] = etok[row0]; }
        }
        const int tb = wid * 8;
        #pragma unroll 1
        for (int nt = 0; nt < 8; ++nt) {
            const int nto = tb + nt;            // output tile [0,64)
            const bf16* po = w2e + ((size_t)(nto * KT2) * 32 + lane) * 16;
            const bf16* arow = is + ln * IDIM;
            v8f acc = {};

            auto loadA = [&](BFrag& A, int kt) {
                const int c0 = kt * 4 + hsel;
                A.q[0] = *(const uint4*)(arow + (((c0)     ^ ln) << 3));
                A.q[1] = *(const uint4*)(arow + (((c0 + 2) ^ ln) << 3));
            };
            auto loadB = [&](BFrag& B, const bf16*& p) {
                B.q[0] = *(const uint4*)(p);
                B.q[1] = *(const uint4*)(p + 8);
                p += 512;
            };

            BFrag A0, B0, A1, B1;
            loadA(A0, 0); loadB(B0, po);
            #pragma unroll 1
            for (int kt = 0; kt < KT2 - 2; kt += 2) {
                loadA(A1, kt + 1); loadB(B1, po);
                acc = WMMA_BF16(A0.v, B0.v, acc);
                loadA(A0, kt + 2); loadB(B0, po);
                acc = WMMA_BF16(A1.v, B1.v, acc);
            }
            loadA(A1, KT2 - 1); loadB(B1, po);
            acc = WMMA_BF16(A0.v, B0.v, acc);
            acc = WMMA_BF16(A1.v, B1.v, acc);

            // exactly TOP_K=2 atomic adds per output element -> order-safe
            const int c = nto * 16 + ln;
            #pragma unroll
            for (int r = 0; r < 8; ++r) {
                atomicAdd(out + (size_t)tk[r] * HDIM + c, acc[r] * sc[r]);
            }
        }
    }
}

// ---------------------------------------------------------------------------
extern "C" void kernel_launch(void* const* d_in, const int* in_sizes, int n_in,
                              void* d_out, int out_size, void* d_ws, size_t ws_size,
                              hipStream_t stream) {
    const float* x      = (const float*)d_in[0];   // [T, H]
    const float* logits = (const float*)d_in[1];   // [T, E]
    const float* w13    = (const float*)d_in[2];   // [E, H, 2I]
    const float* w2     = (const float*)d_in[3];   // [E, I, H]
    float* out = (float*)d_out;                    // [T, H]

    // workspace layout (all 256B-aligned):
    //   counts[8] | btok[E*T] | bw[E*T] | xbf[T*H] | w13f[E*H*2I] | w2f[E*I*H]
    char* ws = (char*)d_ws;
    size_t off = 0;
    int*   counts = (int*)(ws + off);   off += 256;
    int*   btok   = (int*)(ws + off);   off += (size_t)NEXP * T_TOK * sizeof(int);
    float* bw     = (float*)(ws + off); off += (size_t)NEXP * T_TOK * sizeof(float);
    bf16*  xbf    = (bf16*)(ws + off);  off += (size_t)T_TOK * HDIM * sizeof(bf16);
    bf16*  w13f   = (bf16*)(ws + off);  off += (size_t)NEXP * HDIM * N2 * sizeof(bf16);
    bf16*  w2f    = (bf16*)(ws + off);  off += (size_t)NEXP * IDIM * HDIM * sizeof(bf16);

    moe_zero<<<(T_TOK * HDIM + 255) / 256, 256, 0, stream>>>(out, counts);

    const int xdw = (T_TOK * HDIM) / 2;
    moe_conv_x<<<(xdw + 255) / 256, 256, 0, stream>>>(x, (unsigned*)xbf, xdw);

    const int w13dw = (HDIM * N2) / 2;            // per-expert dwords
    moe_conv_w<<<dim3((w13dw + 255) / 256, NEXP), 256, 0, stream>>>(
        w13, (unsigned*)w13f, N2);
    const int w2dw = (IDIM * HDIM) / 2;
    moe_conv_w<<<dim3((w2dw + 255) / 256, NEXP), 256, 0, stream>>>(
        w2, (unsigned*)w2f, HDIM);

    moe_route<<<(T_TOK + 255) / 256, 256, 0, stream>>>(logits, counts, btok, bw);

    dim3 grid(T_TOK / 16, NEXP);   // worst case: all tokens on one expert
    moe_expert<<<grid, 256, 0, stream>>>(xbf, w13f, w2f, counts, btok, bw, out);
}
